// Model_21887153340943
// MI455X (gfx1250) — compile-verified, matrix operand order checked by reading, not run on hardware
//
#include <hip/hip_runtime.h>
#include <cstdint>

typedef _Float16 f16;
typedef _Float16 v16h __attribute__((ext_vector_type(16)));
typedef _Float16 v8h  __attribute__((ext_vector_type(8)));
typedef float    v8f  __attribute__((ext_vector_type(8)));

#define GG 64
#define NN 256
#define FF 20
#define HH 32
#define KK 5
#define NSAMP 100

// ---------------- RNG (deterministic hash; forward-only) ----------------
__device__ __forceinline__ unsigned mix32(unsigned x) {
  x ^= x >> 16; x *= 0x7feb352dU; x ^= x >> 15; x *= 0x846ca68bU; x ^= x >> 16;
  return x;
}
__device__ __forceinline__ unsigned hash4(unsigned a, unsigned b, unsigned c, unsigned d) {
  unsigned h = a * 0x9E3779B1u + 0x165667B1u;
  h = (h ^ b) * 0x85EBCA77u;
  h = (h ^ c) * 0xC2B2AE3Du;
  h = (h ^ d) * 0x27D4EB2Fu;
  return mix32(h);
}
__device__ __forceinline__ float u01(unsigned h) {
  float u = (h >> 8) * 5.9604645e-08f;
  return fminf(fmaxf(u, 1e-6f), 1.f - 1e-6f);
}

// WMMA 16-bit operand fragment: for a lane, elements are two contiguous runs of 8
// halves at K offsets kb and kb+16 (kb = (lane>=16)*8).  Two 128-bit loads.
__device__ __forceinline__ v16h frag(const f16* base, int kb) {
  v8h lo = *(const v8h*)(base + kb);
  v8h hi = *(const v8h*)(base + kb + 16);
  v16h r;
#pragma unroll
  for (int e = 0; e < 8; e++) { r[e] = lo[e]; r[e + 8] = hi[e]; }
  return r;
}
// scalar-gather K index (only used for strided one-time Wtheta operand)
__device__ __forceinline__ int kmap(int e, int kb) {
  int v = e >> 1;
  int base = (v < 4) ? (v << 1) : (16 + ((v - 4) << 1));
  return kb + base + (e & 1);
}
__device__ __forceinline__ v8f wmma16(v16h a, v16h b, v8f c) {
  return __builtin_amdgcn_wmma_f32_16x16x32_f16(false, a, false, b, (short)0, c, false, false);
}

// ---------------- one-time kernels ----------------
__global__ void vgae_init_loss(float* loss) { *loss = 0.f; }

// pre-convert decoder weights to transposed f16 (K-contiguous per output column)
__global__ void vgae_prep(const float* __restrict__ Wd1, const float* __restrict__ Wd2,
                          f16* __restrict__ Wd1T, f16* __restrict__ Wd2T) {
  int idx = blockIdx.x * 256 + threadIdx.x;
  if (idx < HH * HH) {
    int n = idx >> 5, h = idx & 31;
    Wd1T[n * HH + h] = (f16)Wd1[h * HH + n];
    Wd2T[n * HH + h] = (n < FF) ? (f16)Wd2[h * FF + n] : (f16)0.f;
  }
}

__global__ void vgae_encoder(const float* __restrict__ X,
                             const float* __restrict__ Wz1, const float* __restrict__ bz1,
                             const float* __restrict__ Wz2, const float* __restrict__ bz2,
                             const float* __restrict__ Wa1, const float* __restrict__ ba1,
                             float* __restrict__ Zmu, float* __restrict__ ha32,
                             f16* __restrict__ ha16) {
  int g = blockIdx.x, i = threadIdx.x;
  float x[FF];
  const float* xp = X + ((size_t)g * NN + i) * FF;
#pragma unroll
  for (int f = 0; f < FF; f++) x[f] = xp[f];
  float hz[HH];
#pragma unroll 4
  for (int h = 0; h < HH; h++) {
    float s = bz1[h];
#pragma unroll
    for (int f = 0; f < FF; f++) s += x[f] * Wz1[f * HH + h];
    hz[h] = fmaxf(s, 0.f);
  }
  float* zo = Zmu + ((size_t)g * NN + i) * HH;
#pragma unroll 4
  for (int o = 0; o < HH; o++) {
    float s = bz2[o];
#pragma unroll 8
    for (int h = 0; h < HH; h++) s += hz[h] * Wz2[h * HH + o];
    zo[o] = s;
  }
#pragma unroll 4
  for (int h = 0; h < HH; h++) {
    float s = ba1[h];
#pragma unroll
    for (int f = 0; f < FF; f++) s += x[f] * Wa1[f * HH + h];
    s = fmaxf(s, 0.f);
    ha32[((size_t)g * NN + i) * HH + h] = s;
    ha16[((size_t)g * NN + i) * HH + h] = (f16)s;
  }
}

__global__ void vgae_alpha(const float* __restrict__ ha32,
                           const float* __restrict__ Walpha, const float* __restrict__ balpha,
                           float* __restrict__ alpha_out) {
  int g = blockIdx.x, h = threadIdx.x;
  __shared__ float mh[HH];
  float s = 0.f;
  for (int i = 0; i < NN; i++) s += ha32[((size_t)g * NN + i) * HH + h];
  mh[h] = s * (1.f / NN);
  __syncthreads();
  if (h < KK) {
    float a = balpha[h];
#pragma unroll 8
    for (int hh = 0; hh < HH; hh++) a += mh[hh] * Walpha[hh * KK + h];
    alpha_out[g * KK + h] = a;
  }
}

// theta: per (g, mtile): Tk = ha_strip(16x32) @ Wtheta_k(32x32), then logit = Tk @ ha^T
__global__ void vgae_theta(const f16* __restrict__ ha16, const float* __restrict__ Wtheta,
                           float* __restrict__ theta_out) {
  int mt = blockIdx.x, g = blockIdx.y, tid = threadIdx.x;
  __shared__ f16 has[16 * HH];
  __shared__ f16 Tk[KK * 16 * HH];
  for (int idx = tid; idx < 16 * HH; idx += 256) {
    int m = idx >> 5, hh = idx & 31;
    has[idx] = ha16[((size_t)g * NN + mt * 16 + m) * HH + hh];
  }
  __syncthreads();
  int wave = tid >> 5, lane = tid & 31;
  int row = lane & 15, nl = lane & 15, kb = (lane >> 4) * 8, hi = lane >> 4;
  // phase 1: 10 tiles (5 k x 2 column-halves) of Tk   (Wtheta is K-strided; one-time gather)
  for (int t = wave; t < 2 * KK; t += 8) {
    int k = t >> 1, half = t & 1;
    int f = half * 16 + nl;
    v16h a = frag(&has[row * HH], kb);
    v16h b;
#pragma unroll
    for (int e = 0; e < 16; e++) b[e] = (f16)Wtheta[(kmap(e, kb) * HH + f) * KK + k];
    v8f c = {};
    c = wmma16(a, b, c);
#pragma unroll
    for (int r = 0; r < 8; r++) Tk[k * (16 * HH) + (r + hi * 8) * HH + f] = (f16)c[r];
  }
  __syncthreads();
  // phase 2: 80 tiles (5 k x 16 n-tiles) of logits; K=32 -> single WMMA each
  for (int t = wave; t < 16 * KK; t += 8) {
    int k = t / 16, nt = t % 16;
    int j = nt * 16 + nl;
    v16h a = frag(&Tk[k * (16 * HH) + row * HH], kb);
    v16h b = frag(&ha16[((size_t)g * NN + j) * HH], kb);
    v8f c = {};
    c = wmma16(a, b, c);
#pragma unroll
    for (int r = 0; r < 8; r++) {
      int i = mt * 16 + r + hi * 8;
      theta_out[(((size_t)g * NN + i) * NN + j) * KK + k] = c[r];
    }
  }
}

// ---------------- per-sample kernels ----------------
__global__ void vgae_sample(const int* __restrict__ seedp, int s,
                            const float* __restrict__ alpha_out, const float* __restrict__ Zmu,
                            f16* __restrict__ ZsT, int* __restrict__ kstar) {
  int g = blockIdx.x, tid = threadIdx.x;
  unsigned seed = (unsigned)seedp[0];
  __shared__ float av[KK];
  if (tid < KK) {
    float u = u01(hash4(seed, (unsigned)s, (unsigned)g, 0x30000u + tid));
    float gum = -__logf(fmaxf(-__logf(u), 1e-20f));
    av[tid] = alpha_out[g * KK + tid] + gum;
  }
  __syncthreads();
  if (tid == 0) {
    int best = 0; float bv = av[0];
#pragma unroll
    for (int k = 1; k < KK; k++) if (av[k] > bv) { bv = av[k]; best = k; }
    kstar[g] = best;
  }
  const float stdv = 0.22313016f;  // exp(0.5 * Z_LOGVAR) = exp(-1.5)
  int i = tid;
  const float* zp = Zmu + ((size_t)g * NN + i) * HH;
  unsigned c = ((unsigned)g << 8) | (unsigned)i;
#pragma unroll 4
  for (int h = 0; h < HH; h++) {
    float u1 = u01(hash4(seed, (unsigned)s, c, 0x10000u + 2 * h));
    float u2 = u01(hash4(seed, (unsigned)s, c, 0x10000u + 2 * h + 1));
    float n = sqrtf(-2.f * __logf(u1)) * __cosf(6.2831853f * u2);
    ZsT[((size_t)g * HH + h) * NN + i] = (f16)(zp[h] + stdv * n);  // transposed [G][H][N]
  }
}

// regenerate binary symmetric A strip (16 x 256) from hash RNG + selected theta component
__device__ __forceinline__ void gen_A_strip(f16* As, const float* __restrict__ theta,
                                            int g, int mt, int ks, unsigned seed, int s,
                                            int tid, int nthreads) {
  for (int idx = tid; idx < 16 * NN; idx += nthreads) {
    int m = idx >> 8, j = idx & 255;
    int i = mt * 16 + m;
    f16 val = (f16)0.f;
    if (i != j) {
      int r = i > j ? i : j, c2 = i > j ? j : i;  // strict lower-tri coordinate
      float lg = theta[(((size_t)g * NN + r) * NN + c2) * KK + ks];
      float u = u01(hash4(seed, (unsigned)s,
                          ((unsigned)g << 16) | ((unsigned)r << 8) | (unsigned)c2, 0x20000u));
      float logistic = __logf(u) - __logf(1.f - u);
      val = ((lg + logistic) > 0.f) ? (f16)1.f : (f16)0.f;  // hard binary-concrete forward
    }
    As[m * NN + j] = val;
  }
}

// phase 1 GCN: h = relu((A @ Zs) @ Wd1 + bd1);  h stored transposed [G][H][N]
__global__ void vgae_gcn1(const float* __restrict__ theta, const f16* __restrict__ ZsT,
                          const f16* __restrict__ Wd1T, const float* __restrict__ bd1,
                          f16* __restrict__ hT, const int* __restrict__ kstar,
                          const int* __restrict__ seedp, int s) {
  int mt = blockIdx.x, g = blockIdx.y, tid = threadIdx.x;
  __shared__ f16 As[16 * NN];
  __shared__ f16 AZ[16 * HH];
  unsigned seed = (unsigned)seedp[0];
  int ks = kstar[g];
  gen_A_strip(As, theta, g, mt, ks, seed, s, tid, 64);
  __syncthreads();
  int wave = tid >> 5, lane = tid & 31;
  int row = lane & 15, nl = lane & 15, kb = (lane >> 4) * 8, hi = lane >> 4;
  int n = wave * 16 + nl;
  v8f c = {};
  const f16* zrow = ZsT + ((size_t)g * HH + n) * NN;
  for (int kt = 0; kt < 8; kt++) {  // K = 256 in 8 steps of 32
    v16h a = frag(&As[row * NN + kt * 32], kb);
    v16h b = frag(zrow + kt * 32, kb);
    c = wmma16(a, b, c);
  }
#pragma unroll
  for (int r = 0; r < 8; r++) AZ[(r + hi * 8) * HH + n] = (f16)c[r];
  __syncthreads();
  v16h a = frag(&AZ[row * HH], kb);
  v16h b = frag(&Wd1T[n * HH], kb);
  v8f c2 = {};
  c2 = wmma16(a, b, c2);
  float bn = bd1[n];
#pragma unroll
  for (int r = 0; r < 8; r++) {
    float hv = fmaxf(c2[r] + bn, 0.f);
    hT[((size_t)g * HH + n) * NN + mt * 16 + r + hi * 8] = (f16)hv;
  }
}

// phase 2 GCN: Xmu = (A @ h) @ Wd2 + bd2 ; accumulate 0.5*sum((Xmu-X)^2)/(G*NSAMP)
__global__ void vgae_gcn2(const float* __restrict__ theta, const f16* __restrict__ hT,
                          const f16* __restrict__ Wd2T, const float* __restrict__ bd2,
                          const float* __restrict__ X, float* __restrict__ loss,
                          const int* __restrict__ kstar, const int* __restrict__ seedp, int s) {
  int mt = blockIdx.x, g = blockIdx.y, tid = threadIdx.x;
  __shared__ f16 As[16 * NN];
  __shared__ f16 Ah[16 * HH];
  __shared__ float red[64];
  unsigned seed = (unsigned)seedp[0];
  int ks = kstar[g];
  gen_A_strip(As, theta, g, mt, ks, seed, s, tid, 64);
  __syncthreads();
  int wave = tid >> 5, lane = tid & 31;
  int row = lane & 15, nl = lane & 15, kb = (lane >> 4) * 8, hi = lane >> 4;
  int n = wave * 16 + nl;
  v8f c = {};
  const f16* hrow = hT + ((size_t)g * HH + n) * NN;
  for (int kt = 0; kt < 8; kt++) {
    v16h a = frag(&As[row * NN + kt * 32], kb);
    v16h b = frag(hrow + kt * 32, kb);
    c = wmma16(a, b, c);
  }
#pragma unroll
  for (int r = 0; r < 8; r++) Ah[(r + hi * 8) * HH + n] = (f16)c[r];
  __syncthreads();
  v16h a = frag(&Ah[row * HH], kb);
  v16h b = frag(&Wd2T[n * HH], kb);  // rows n>=20 pre-zeroed
  v8f c2 = {};
  c2 = wmma16(a, b, c2);
  float acc = 0.f;
  if (n < FF) {
    float bn = bd2[n];
#pragma unroll
    for (int r = 0; r < 8; r++) {
      int i = mt * 16 + r + hi * 8;
      float d = (c2[r] + bn) - X[((size_t)g * NN + i) * FF + n];
      acc += d * d;
    }
  }
  red[tid] = acc;
  __syncthreads();
  if (tid == 0) {
    float sum = 0.f;
#pragma unroll 8
    for (int t = 0; t < 64; t++) sum += red[t];
    atomicAdd(loss, sum * (0.5f / ((float)GG * (float)NSAMP)));
  }
}

// in-place: prob_theta = sigmoid(logit) * offdiag-mask (runs last; samplers read raw logits)
__global__ void vgae_sigmoid(float* __restrict__ theta) {
  size_t idx = (size_t)blockIdx.x * 256 + threadIdx.x;
  size_t t = idx / KK;
  int j = (int)(t % NN);
  int i = (int)((t / NN) % NN);
  float x = theta[idx];
  float p = 1.f / (1.f + __expf(-x));
  theta[idx] = (i == j) ? 0.f : p;
}

extern "C" void kernel_launch(void* const* d_in, const int* in_sizes, int n_in,
                              void* d_out, int out_size, void* d_ws, size_t ws_size,
                              hipStream_t stream) {
  const float* X      = (const float*)d_in[0];
  const float* Wz1    = (const float*)d_in[1];
  const float* bz1    = (const float*)d_in[2];
  const float* Wz2    = (const float*)d_in[3];
  const float* bz2    = (const float*)d_in[4];
  const float* Wa1    = (const float*)d_in[5];
  const float* ba1    = (const float*)d_in[6];
  const float* Wtheta = (const float*)d_in[7];
  const float* Walpha = (const float*)d_in[8];
  const float* balpha = (const float*)d_in[9];
  const float* Wd1    = (const float*)d_in[10];
  const float* bd1    = (const float*)d_in[11];
  const float* Wd2    = (const float*)d_in[12];
  const float* bd2    = (const float*)d_in[13];
  const int*   seed   = (const int*)d_in[14];

  float* out   = (float*)d_out;
  float* theta = out;                                   // [G,N,N,K] = 20,971,520
  float* alpha = out + (size_t)GG * NN * NN * KK;       // [G,K]     = 320
  float* Zmu   = alpha + (size_t)GG * KK;               // [G,N,H]   = 524,288
  float* loss  = Zmu + (size_t)GG * NN * HH;            // scalar

  char* wsb   = (char*)d_ws;
  float* ha32 = (float*)wsb;                            // 2 MB
  f16* ha16   = (f16*)(wsb + 2097152);                  // 1 MB
  f16* ZsT    = (f16*)(wsb + 3145728);                  // 1 MB  [G][H][N]
  f16* hT     = (f16*)(wsb + 4194304);                  // 1 MB  [G][H][N]
  int* kstar  = (int*)(wsb + 5242880);                  // 256 B
  f16* Wd1T   = (f16*)(wsb + 5243136);                  // 2 KB
  f16* Wd2T   = (f16*)(wsb + 5245184);                  // 2 KB

  vgae_init_loss<<<1, 1, 0, stream>>>(loss);
  vgae_prep<<<4, 256, 0, stream>>>(Wd1, Wd2, Wd1T, Wd2T);
  vgae_encoder<<<GG, 256, 0, stream>>>(X, Wz1, bz1, Wz2, bz2, Wa1, ba1, Zmu, ha32, ha16);
  vgae_alpha<<<GG, HH, 0, stream>>>(ha32, Walpha, balpha, alpha);
  vgae_theta<<<dim3(16, GG), 256, 0, stream>>>(ha16, Wtheta, theta);
  for (int s = 0; s < NSAMP; s++) {
    vgae_sample<<<GG, 256, 0, stream>>>(seed, s, alpha, Zmu, ZsT, kstar);
    vgae_gcn1<<<dim3(16, GG), 64, 0, stream>>>(theta, ZsT, Wd1T, bd1, hT, kstar, seed, s);
    vgae_gcn2<<<dim3(16, GG), 64, 0, stream>>>(theta, hT, Wd2T, bd2, X, loss, kstar, seed, s);
  }
  vgae_sigmoid<<<(GG * NN * NN * KK) / 256, 256, 0, stream>>>(theta);
}
